// SwinTinyEncoder_44238163149160
// MI455X (gfx1250) — matrix operand order repaired and checked by reading, not compiled, not run on hardware
//
#include <hip/hip_runtime.h>
#include <hip/hip_bf16.h>
#include <math.h>

// ---------------------------------------------------------------------------
// Swin-Tiny forward for MI455X (gfx1250, wave32, WMMA).
// - All GEMMs: v_wmma_f32_16x16x32_bf16, weights pre-converted to bf16 once.
// - Register-pipelined global->LDS staging (no exec-diverged guarded loads).
// - Window attention fully fused in LDS; wave-per-row softmax.
// ---------------------------------------------------------------------------

typedef __attribute__((ext_vector_type(16))) __bf16 v16bf;
typedef __attribute__((ext_vector_type(8)))  float  v8f;

#define WMMA_BF16(a, b, c) \
  __builtin_amdgcn_wmma_f32_16x16x32_bf16(false, (a), false, (b), (short)0, (c), false, false)

static __device__ __forceinline__ __bf16 f2bf(float x) { return (__bf16)x; }

static __device__ __forceinline__ float geluf(float x) {
  return 0.5f * x * (1.0f + erff(x * 0.70710678118654752f));
}

// ---------------------------------------------------------------------------
// fp32 -> bf16 bulk conversion (weights, once per launch)
// ---------------------------------------------------------------------------
__global__ void k_cvt_bf16(const float* __restrict__ in, __bf16* __restrict__ out, int n) {
  int idx = (blockIdx.x * blockDim.x + threadIdx.x) * 4;
  if (idx + 3 < n) {
    float4 v = *(const float4*)(in + idx);
    out[idx] = f2bf(v.x); out[idx + 1] = f2bf(v.y);
    out[idx + 2] = f2bf(v.z); out[idx + 3] = f2bf(v.w);
  } else {
    for (int j = idx; j < n; j++) out[j] = f2bf(in[j]);
  }
}

// ---------------------------------------------------------------------------
// Patch embed: 4x4 stride-4 conv, NCHW input -> (B,56,56,96) NHWC + bias
// ---------------------------------------------------------------------------
__global__ void k_patch_embed(const float* __restrict__ img,
                              const float* __restrict__ w,
                              const float* __restrict__ bias,
                              float* __restrict__ out) {
  int p  = blockIdx.x;            // b*56*56 + oh*56 + ow
  int oc = threadIdx.x;           // 0..95
  int ow = p % 56; int t = p / 56; int oh = t % 56; int b = t / 56;
  float acc = bias[oc];
  for (int ci = 0; ci < 3; ci++)
    for (int kh = 0; kh < 4; kh++)
      for (int kw = 0; kw < 4; kw++)
        acc += img[(((size_t)b * 3 + ci) * 224 + (oh * 4 + kh)) * 224 + (ow * 4 + kw)] *
               w[((oc * 3 + ci) * 4 + kh) * 4 + kw];
  out[(size_t)p * 96 + oc] = acc;
}

// ---------------------------------------------------------------------------
// LayerNorm over last dim C; one wave32 per row (in-place safe).
// ---------------------------------------------------------------------------
__global__ __launch_bounds__(256) void k_layernorm(const float* __restrict__ x,
                                                   const float* __restrict__ g,
                                                   const float* __restrict__ b,
                                                   float* __restrict__ y,
                                                   int M, int C) {
  int wave = threadIdx.x >> 5, lane = threadIdx.x & 31;
  int row = blockIdx.x * 8 + wave;
  if (row >= M) return;
  const float* xr = x + (size_t)row * C;
  float s = 0.f, s2 = 0.f;
  for (int c = lane; c < C; c += 32) { float v = xr[c]; s += v; s2 += v * v; }
  for (int o = 16; o; o >>= 1) { s += __shfl_xor(s, o, 32); s2 += __shfl_xor(s2, o, 32); }
  float mean = s / C;
  float var  = s2 / C - mean * mean;
  float inv  = rsqrtf(var + 1e-5f);
  float* yr = y + (size_t)row * C;
  for (int c = lane; c < C; c += 32) yr[c] = (xr[c] - mean) * inv * g[c] + b[c];
}

// ---------------------------------------------------------------------------
// Window partition with optional cyclic shift: (B,H,W,C) -> (B*nW*49, C)
// ---------------------------------------------------------------------------
__global__ void k_partition(const float* __restrict__ x, float* __restrict__ xw,
                            int B, int Hc, int C, int shift) {
  size_t idx = (size_t)blockIdx.x * blockDim.x + threadIdx.x;
  size_t total = (size_t)B * Hc * Hc * C;
  if (idx >= total) return;
  int c = (int)(idx % C); size_t row = idx / C;
  int n = (int)(row % 49); size_t wI = row / 49;
  int Wg = Hc / 7;
  int wi = (int)(wI % (Wg * Wg)); int b = (int)(wI / (Wg * Wg));
  int wh = wi / Wg, ww = wi % Wg;
  int r = n / 7, cc = n % 7;
  int hh = (wh * 7 + r + shift) % Hc;
  int wp = (ww * 7 + cc + shift) % Hc;
  xw[idx] = x[(((size_t)b * Hc + hh) * Hc + wp) * C + c];
}

// Reverse windows (+unshift) and add residual into x.
__global__ void k_reverse_add(float* __restrict__ x, const float* __restrict__ pw,
                              int B, int Hc, int C, int shift) {
  size_t idx = (size_t)blockIdx.x * blockDim.x + threadIdx.x;
  size_t total = (size_t)B * Hc * Hc * C;
  if (idx >= total) return;
  int c = (int)(idx % C); size_t t = idx / C;
  int wcol = (int)(t % Hc); size_t t2 = t / Hc;
  int hrow = (int)(t2 % Hc); int b = (int)(t2 / Hc);
  int h2 = (hrow - shift + Hc) % Hc;
  int w2 = (wcol - shift + Hc) % Hc;
  int Wg = Hc / 7;
  int wh = h2 / 7, r = h2 % 7, ww = w2 / 7, cc = w2 % 7;
  size_t row = (((size_t)b * Wg + wh) * Wg + ww) * 49 + r * 7 + cc;
  x[idx] += pw[row * C + c];
}

// Patch merging gather: (B,H,W,C) -> (B*(H/2)*(W/2), 4C)
__global__ void k_merge(const float* __restrict__ x, float* __restrict__ y,
                        int B, int Hc, int C) {
  size_t idx = (size_t)blockIdx.x * blockDim.x + threadIdx.x;
  int H2 = Hc / 2;
  size_t total = (size_t)B * H2 * H2 * 4 * C;
  if (idx >= total) return;
  int c4 = (int)(idx % (4 * C)); size_t t = idx / (4 * C);
  int j = (int)(t % H2); size_t t2 = t / H2;
  int i = (int)(t2 % H2); int b = (int)(t2 / H2);
  int q = c4 / C, c = c4 % C;
  int di = (q == 1 || q == 3) ? 1 : 0;  // x1,x3 take odd rows
  int dj = (q >= 2) ? 1 : 0;            // x2,x3 take odd cols
  y[idx] = x[(((size_t)b * Hc + 2 * i + di) * Hc + 2 * j + dj) * C + c];
}

// qkv bias with k-bias zeroed (reference: qkv_b[C:2C] = 0)
__global__ void k_qkvbias(const float* __restrict__ qb, float* __restrict__ ob, int C) {
  int i = blockIdx.x * blockDim.x + threadIdx.x;
  if (i < 3 * C) ob[i] = (i >= C && i < 2 * C) ? 0.f : qb[i];
}

// Final mean over 49 tokens per image: (32*49,768) -> (32,768)
__global__ void k_meanpool(const float* __restrict__ y, float* __restrict__ out) {
  int i = blockIdx.x * blockDim.x + threadIdx.x;
  if (i >= 32 * 768) return;
  int b = i / 768, c = i % 768;
  float s = 0.f;
  for (int n = 0; n < 49; n++) s += y[((size_t)b * 49 + n) * 768 + c];
  out[i] = s * (1.0f / 49.0f);
}

// ---------------------------------------------------------------------------
// WMMA GEMM: out[M,N] = act(A[M,K](f32) @ W[N,K](bf16)^T + bias) (+ res)
// 256 threads = 8 wave32; tile 128x64, K-step 32.
// Register-pipelined staging: fetch k+1 while computing k. Row-clamped loads
// (out-of-range rows feed only masked outputs) keep the fill branch-free.
// ---------------------------------------------------------------------------
#define BM 128
#define BN 64
#define BK 32
#define LDA 40  // row stride (bf16 elems): 80B, keeps fragments 16B-aligned

__global__ __launch_bounds__(256) void k_gemm(const float* __restrict__ A,
                                              const __bf16* __restrict__ W,
                                              const float* __restrict__ bias,
                                              const float* __restrict__ res,
                                              float* __restrict__ out,
                                              int M, int N, int K, int act) {
  __shared__ __align__(16) __bf16 As[BM][LDA];
  __shared__ __align__(16) __bf16 Ws[BN][LDA];
  const int tid  = threadIdx.x;
  const int lane = tid & 31;
  const int wave = tid >> 5;
  const int bm = blockIdx.x * BM;
  const int bn = blockIdx.y * BN;
  const int wm = (wave & 3) * 32;   // 4 waves along M
  const int wn = (wave >> 2) * 32;  // 2 waves along N
  const int mlane = lane & 15;
  const int kbA = (lane < 16) ? 0 : 8;   // unused; A frag reads K 0-15 vs 16-31 via kbB
  const int kbB = (lane < 16) ? 0 : 16;  // B frag: K 0-15 vs 16-31

  float4 aR[4];
  uint4  wR;

  auto fetch = [&](int k0) {
#pragma unroll
    for (int t = 0; t < 4; t++) {
      int idx = tid + (t << 8);
      int r = idx >> 3, c = (idx & 7) << 2;
      int gr = bm + r; gr = (gr < M) ? gr : (M - 1);
      aR[t] = *(const float4*)(A + (size_t)gr * K + k0 + c);
    }
    {
      int r = tid >> 2, c = (tid & 3) << 3;
      int gr = bn + r; gr = (gr < N) ? gr : (N - 1);
      wR = *(const uint4*)(W + (size_t)gr * K + k0 + c);
    }
  };
  auto stage = [&]() {
#pragma unroll
    for (int t = 0; t < 4; t++) {
      int idx = tid + (t << 8);
      int r = idx >> 3, c = (idx & 7) << 2;
      As[r][c] = f2bf(aR[t].x); As[r][c + 1] = f2bf(aR[t].y);
      As[r][c + 2] = f2bf(aR[t].z); As[r][c + 3] = f2bf(aR[t].w);
    }
    {
      int r = tid >> 2, c = (tid & 3) << 3;
      *(uint4*)&Ws[r][c] = wR;
    }
  };

  v8f acc[2][2] = {};
  fetch(0);
  for (int k0 = 0; k0 < K; k0 += BK) {
    stage();
    __syncthreads();
    if (k0 + BK < K) fetch(k0 + BK);  // overlap next-tile loads with WMMA

    v16bf bf[2];
#pragma unroll
    for (int j = 0; j < 2; j++) {
      int n = wn + j * 16 + mlane;
#pragma unroll
      for (int i = 0; i < 16; i++) bf[j][i] = Ws[n][kbB + i];
    }
#pragma unroll
    for (int ii = 0; ii < 2; ii++) {
      v16bf af;
      int m = wm + ii * 16 + mlane;
#pragma unroll
      for (int i = 0; i < 16; i++) af[i] = As[m][kbB + i];
      acc[ii][0] = WMMA_BF16(af, bf[0], acc[ii][0]);
      acc[ii][1] = WMMA_BF16(af, bf[1], acc[ii][1]);
    }
    __syncthreads();
  }

  const int colL = lane & 15;
  const int rowOff = (lane < 16) ? 0 : 8;  // D: lanes 0-15 rows 0-7, 16-31 rows 8-15
  for (int ii = 0; ii < 2; ii++)
    for (int j = 0; j < 2; j++) {
#pragma unroll
      for (int r = 0; r < 8; r++) {
        int gm = bm + wm + ii * 16 + rowOff + r;
        int gn = bn + wn + j * 16 + colL;
        if (gm < M && gn < N) {
          float v = acc[ii][j][r];
          if (bias) v += bias[gn];
          if (act)  v = geluf(v);
          if (res)  v += res[(size_t)gm * N + gn];
          out[(size_t)gm * N + gn] = v;
        }
      }
    }
}

// ---------------------------------------------------------------------------
// Fused window attention, one workgroup per (window, head). head_dim = 32.
// QK^T (WMMA) -> +rel-bias +shift-mask -> wave-per-row softmax -> P@V (WMMA).
// V stored transposed in LDS so the P@V B-fragment is a contiguous read.
// ---------------------------------------------------------------------------
__global__ __launch_bounds__(256) void k_attn(const float* __restrict__ qkv,
                                              const float* __restrict__ relb,
                                              float* __restrict__ out,
                                              int nh, int C, int nWimg, int Wg,
                                              int Hc, int shift) {
  const int w = blockIdx.x;
  const int h = blockIdx.y;
  const int tid = threadIdx.x;
  const int lane = tid & 31, wave = tid >> 5;

  __shared__ __align__(16) __bf16 qs[64][40];
  __shared__ __align__(16) __bf16 ks[64][40];
  __shared__ __align__(16) __bf16 vsT[32][72];  // transposed V: [d][row]
  __shared__ float  Sm[64][66];
  __shared__ __align__(16) __bf16 Ps[64][72];

  const float scale = 0.17677669529663687f;  // 32^-0.5
  const size_t base = (size_t)w * 49 * (3 * C) + (size_t)h * 32;
  for (int i = tid; i < 64 * 8; i += 256) {  // 2 iters, branch-free (row clamp)
    int r = i >> 3, d4 = (i & 7) << 2;
    int rc = (r < 49) ? r : 48;
    const float* p = qkv + base + (size_t)rc * (3 * C) + d4;
    float4 q4 = *(const float4*)(p);
    float4 k4 = *(const float4*)(p + C);
    float4 v4 = *(const float4*)(p + 2 * C);
    qs[r][d4 + 0] = f2bf(q4.x * scale); qs[r][d4 + 1] = f2bf(q4.y * scale);
    qs[r][d4 + 2] = f2bf(q4.z * scale); qs[r][d4 + 3] = f2bf(q4.w * scale);
    ks[r][d4 + 0] = f2bf(k4.x); ks[r][d4 + 1] = f2bf(k4.y);
    ks[r][d4 + 2] = f2bf(k4.z); ks[r][d4 + 3] = f2bf(k4.w);
    vsT[d4 + 0][r] = f2bf(v4.x); vsT[d4 + 1][r] = f2bf(v4.y);
    vsT[d4 + 2][r] = f2bf(v4.z); vsT[d4 + 3][r] = f2bf(v4.w);
  }
  __syncthreads();

  // S = q @ k^T : 64x64, single K=32 step. 8 waves x 2 tiles.
  {
    const int wm = (wave & 3) * 16;
    const int wn = (wave >> 2) * 32;
    const int mlane = lane & 15;
    const int kbA = (lane < 16) ? 0 : 8;
    const int kbB = (lane < 16) ? 0 : 16;
    v16bf af;
#pragma unroll
    for (int i = 0; i < 16; i++) af[i] = qs[wm + mlane][kbB + i];
    v8f acc[2] = {};
#pragma unroll
    for (int j = 0; j < 2; j++) {
      v16bf bf;
      int n = wn + j * 16 + mlane;
#pragma unroll
      for (int i = 0; i < 16; i++) bf[i] = ks[n][kbB + i];
      acc[j] = WMMA_BF16(af, bf, acc[j]);
    }
    const int colL = lane & 15;
    const int rowOff = (lane < 16) ? 0 : 8;
#pragma unroll
    for (int j = 0; j < 2; j++)
#pragma unroll
      for (int r = 0; r < 8; r++)
        Sm[wm + rowOff + r][wn + j * 16 + colL] = acc[j][r];
  }
  __syncthreads();

  // Softmax: one wave per row, lanes cover columns {lane, lane+32}.
  for (int row = wave; row < 64; row += 8) {
    if (row < 49) {
      int r1 = row / 7, c1 = row % 7;
      int cn = 0, wh = 0, ww = 0;
      if (shift > 0) {
        int wi = w % nWimg;
        wh = wi / Wg; ww = wi % Wg;
        int gh = wh * 7 + r1, gw = ww * 7 + c1;
        int ih = (gh < Hc - 7) ? 0 : (gh < Hc - shift ? 1 : 2);
        int iw = (gw < Hc - 7) ? 0 : (gw < Hc - shift ? 1 : 2);
        cn = ih * 3 + iw;
      }
      float v0 = -3.0e38f, v1 = -3.0e38f;
      int m0 = lane, m1 = lane + 32;
#pragma unroll
      for (int half = 0; half < 2; half++) {
        int m = half ? m1 : m0;
        if (m < 49) {
          int r2 = m / 7, c2 = m % 7;
          int idx = (r1 - r2 + 6) * 13 + (c1 - c2 + 6);
          float v = Sm[row][m] + relb[idx * nh + h];
          if (shift > 0) {
            int gh = wh * 7 + r2, gw = ww * 7 + c2;
            int ih = (gh < Hc - 7) ? 0 : (gh < Hc - shift ? 1 : 2);
            int iw = (gw < Hc - 7) ? 0 : (gw < Hc - shift ? 1 : 2);
            if (ih * 3 + iw != cn) v -= 100.0f;
          }
          if (half) v1 = v; else v0 = v;
        }
      }
      float mx = fmaxf(v0, v1);
      for (int o = 16; o; o >>= 1) mx = fmaxf(mx, __shfl_xor(mx, o, 32));
      float e0 = (m0 < 49) ? __expf(v0 - mx) : 0.f;
      float e1 = (m1 < 49) ? __expf(v1 - mx) : 0.f;
      float s = e0 + e1;
      for (int o = 16; o; o >>= 1) s += __shfl_xor(s, o, 32);
      float inv = 1.0f / s;
      Ps[row][m0] = f2bf(e0 * inv);
      Ps[row][m1] = f2bf(e1 * inv);
    } else {
      Ps[row][lane] = f2bf(0.f);
      Ps[row][lane + 32] = f2bf(0.f);
    }
  }
  __syncthreads();

  // O = P @ V : 64x32, K=64 (P cols 49..63 are zero). 8 waves x 1 tile.
  {
    const int wm = (wave & 3) * 16;
    const int wn = (wave >> 2) * 16;
    const int mlane = lane & 15;
    const int kbA = (lane < 16) ? 0 : 8;
    const int kbB = (lane < 16) ? 0 : 16;
    v8f acc = {};
#pragma unroll
    for (int k0 = 0; k0 < 64; k0 += 32) {
      v16bf af, bf;
#pragma unroll
      for (int i = 0; i < 16; i++) af[i] = Ps[wm + mlane][k0 + kbB + i];
#pragma unroll
      for (int i = 0; i < 16; i++) bf[i] = vsT[wn + mlane][k0 + kbB + i];
      acc = WMMA_BF16(af, bf, acc);
    }
    const int colL = lane & 15;
    const int rowOff = (lane < 16) ? 0 : 8;
#pragma unroll
    for (int r = 0; r < 8; r++) {
      int n = wm + rowOff + r;
      int d = wn + colL;
      if (n < 49) out[((size_t)w * 49 + n) * C + h * 32 + d] = acc[r];
    }
  }
}

// ---------------------------------------------------------------------------
// Host orchestration. d_in order = jax pytree flatten (sorted dict keys):
// 0 inputs | 1 ln0_b 2 ln0_g | 3..11 merges(ln_b,ln_g,red_w)x3 |
// 12 norm_b 13 norm_g | 14 patch_b 15 patch_w |
// 16+ blocks x13 sorted: fc1_b fc1_w fc2_b fc2_w ln1_b ln1_g ln2_b ln2_g
//                        proj_b proj_w qkv_b qkv_w rel_bias
// ---------------------------------------------------------------------------
extern "C" void kernel_launch(void* const* d_in, const int* in_sizes, int n_in,
                              void* d_out, int out_size, void* d_ws, size_t ws_size,
                              hipStream_t stream) {
  (void)in_sizes; (void)n_in; (void)out_size; (void)ws_size;
  auto F = [&](int i) { return (const float*)d_in[i]; };
  const int B = 32;

  float* ws  = (float*)d_ws;
  const size_t SZ = 9633792;        // 100352 * 96
  float* X   = ws;                  // residual stream
  float* Y   = X + SZ;              // LN output
  float* XW  = Y + SZ;              // windowed / proj-out / merge-LN out
  float* O   = XW + SZ;             // attention output
  float* BIG = O + SZ;              // QKV and MLP hidden (100352*384)
  float* QB  = BIG + 38535168;      // masked qkv bias (<= 2304 floats)
  __bf16* WB = (__bf16*)(QB + 4096);// bf16 weight arena (~27.5M elems)

  size_t wcur = 0;
  auto prep = [&](int pi, size_t n) -> const __bf16* {
    __bf16* dst = WB + wcur;
    wcur += n;
    k_cvt_bf16<<<(int)((n / 4 + 255) / 256), 256, 0, stream>>>(F(pi), dst, (int)n);
    return dst;
  };

  // Patch embed + LN0 (in-place)
  k_patch_embed<<<32 * 56 * 56, 96, 0, stream>>>(F(0), F(15), F(14), X);
  int M0 = 32 * 56 * 56;
  k_layernorm<<<(M0 + 7) / 8, 256, 0, stream>>>(X, F(2), F(1), X, M0, 96);

  const int Hs[4] = {56, 28, 14, 7};
  const int Cs[4] = {96, 192, 384, 768};
  const int NHs[4] = {3, 6, 12, 24};
  const int DEP[4] = {2, 2, 6, 2};
  const int stageOff[4] = {0, 2, 4, 10};

  for (int s = 0; s < 4; s++) {
    int Hc = Hs[s], C = Cs[s], nh = NHs[s];
    int M = B * Hc * Hc;
    int Wg = Hc / 7;
    int NW = B * Wg * Wg;
    size_t tot = (size_t)M * C;
    for (int blk = 0; blk < DEP[s]; blk++) {
      int pb = 16 + (stageOff[s] + blk) * 13;
      int shift = (blk % 2 == 1 && Hc > 7) ? 3 : 0;

      const __bf16* qkvW = prep(pb + 11, (size_t)3 * C * C);
      const __bf16* projW = prep(pb + 9, (size_t)C * C);
      const __bf16* fc1W = prep(pb + 1, (size_t)4 * C * C);
      const __bf16* fc2W = prep(pb + 3, (size_t)4 * C * C);

      // LN1 -> Y ; window partition (+shift) -> XW
      k_layernorm<<<(M + 7) / 8, 256, 0, stream>>>(X, F(pb + 5), F(pb + 4), Y, M, C);
      k_partition<<<(int)((tot + 255) / 256), 256, 0, stream>>>(Y, XW, B, Hc, C, shift);

      // QKV projection (with k-bias zeroed)
      k_qkvbias<<<(3 * C + 255) / 256, 256, 0, stream>>>(F(pb + 10), QB, C);
      dim3 g1((M + BM - 1) / BM, (3 * C + BN - 1) / BN);
      k_gemm<<<g1, 256, 0, stream>>>(XW, qkvW, QB, nullptr, BIG, M, 3 * C, C, 0);

      // Fused window attention
      dim3 g2(NW, nh);
      k_attn<<<g2, 256, 0, stream>>>(BIG, F(pb + 12), O, nh, C, Wg * Wg, Wg, Hc, shift);

      // Output projection -> XW ; window reverse + residual add -> X
      dim3 g3((M + BM - 1) / BM, (C + BN - 1) / BN);
      k_gemm<<<g3, 256, 0, stream>>>(O, projW, F(pb + 8), nullptr, XW, M, C, C, 0);
      k_reverse_add<<<(int)((tot + 255) / 256), 256, 0, stream>>>(X, XW, B, Hc, C, shift);

      // MLP: LN2 -> fc1(GELU) -> fc2 (+residual fused into X)
      k_layernorm<<<(M + 7) / 8, 256, 0, stream>>>(X, F(pb + 7), F(pb + 6), Y, M, C);
      dim3 g4((M + BM - 1) / BM, (4 * C + BN - 1) / BN);
      k_gemm<<<g4, 256, 0, stream>>>(Y, fc1W, F(pb + 0), nullptr, BIG, M, 4 * C, C, 1);
      dim3 g5((M + BM - 1) / BM, (C + BN - 1) / BN);
      k_gemm<<<g5, 256, 0, stream>>>(BIG, fc2W, F(pb + 2), X, X, M, C, 4 * C, 0);
    }
    if (s < 3) {
      const __bf16* redW = prep(5 + 3 * s, (size_t)8 * C * C);
      int M2 = B * (Hc / 2) * (Hc / 2);
      size_t tot2 = (size_t)M2 * 4 * C;
      k_merge<<<(int)((tot2 + 255) / 256), 256, 0, stream>>>(X, Y, B, Hc, C);
      k_layernorm<<<(M2 + 7) / 8, 256, 0, stream>>>(Y, F(4 + 3 * s), F(3 + 3 * s), XW, M2, 4 * C);
      dim3 g6((M2 + BM - 1) / BM, (2 * C + BN - 1) / BN);
      k_gemm<<<g6, 256, 0, stream>>>(XW, redW, nullptr, nullptr, X, M2, 2 * C, 4 * C, 0);
    }
  }

  // Final LN + mean pool -> (32, 768)
  int Mf = 32 * 49;
  k_layernorm<<<(Mf + 7) / 8, 256, 0, stream>>>(X, F(13), F(12), Y, Mf, 768);
  k_meanpool<<<(32 * 768 + 255) / 256, 256, 0, stream>>>(Y, (float*)d_out);
}
